// POSTagger_84482006713072
// MI455X (gfx1250) — matrix operand order repaired and checked by reading, not compiled, not run on hardware
//
#include <hip/hip_runtime.h>
#include <hip/hip_bf16.h>
#include <stdint.h>

// ---------------- problem constants ----------------
#define SEQ     256
#define BATCH   64
#define EMB     512
#define HID     256
#define NTAGS   50
#define NLAYERS 2
#define G4H     1024   // 4*HID
#define IN2     512    // 2*HID (== EMB, so all GEMMs share K=512)

typedef __bf16 bf16;
typedef __attribute__((ext_vector_type(16))) __bf16 v16bf;
typedef __attribute__((ext_vector_type(8)))  float  v8f;

union FragU { v16bf v; unsigned u[8]; };

// ---------------- gfx1250 async global->LDS path (compile-safe) ------------
#if __has_builtin(__builtin_amdgcn_global_load_async_to_lds_b128) && \
    __has_builtin(__builtin_amdgcn_s_wait_asynccnt)
#define HAVE_ASYNC_LDS 1
typedef int v4i __attribute__((vector_size(16)));
typedef __attribute__((address_space(1))) v4i* gv4p;   // global int4*
typedef __attribute__((address_space(3))) v4i* lv4p;   // LDS int4*
#endif

__device__ __forceinline__ void copy16_g2l(const void* g, void* l) {
#ifdef HAVE_ASYNC_LDS
  __builtin_amdgcn_global_load_async_to_lds_b128((gv4p)g, (lv4p)l, 0, 0);
#else
  *(uint4*)l = *(const uint4*)g;
#endif
}
__device__ __forceinline__ void async_fence_wait() {
#ifdef HAVE_ASYNC_LDS
  __builtin_amdgcn_s_wait_asynccnt(0);
#endif
}

// ---------------- WMMA fragment loaders (CDNA5 ISA 7.12.2 layouts) ---------
// A: 16x32 bf16, row-major source p[m*ldk + k]
__device__ __forceinline__ v16bf frag_a(const unsigned short* p, int mbase, int kbase, int ldk) {
  const int lane  = threadIdx.x & 31;
  const int m     = mbase + (lane & 15);
  const int khalf = (lane & 16) ? 8 : 0;
  FragU f;
#pragma unroll
  for (int v = 0; v < 8; ++v) {
    int k = kbase + khalf + 2 * v + ((v >= 4) ? 8 : 0);
    f.u[v] = *(const unsigned*)(p + (size_t)m * ldk + k);
  }
  return f.v;
}
// B: 32x16 bf16 of W^T, source is W[n*ldk + k] (N x K row-major)
__device__ __forceinline__ v16bf frag_b(const unsigned short* p, int nbase, int kbase, int ldk) {
  const int lane = threadIdx.x & 31;
  const int n    = nbase + (lane & 15);
  const int koff = (lane & 16) ? 8 : 0;
  FragU f;
#pragma unroll
  for (int v = 0; v < 8; ++v) {
    int k = kbase + koff + 2 * v + ((v >= 4) ? 8 : 0);
    f.u[v] = *(const unsigned*)(p + (size_t)n * ldk + k);
  }
  return f.v;
}

__device__ __forceinline__ float sigf(float x) { return 1.0f / (1.0f + __expf(-x)); }
__device__ __forceinline__ unsigned short f2bf_bits(float x) {
  bf16 b = (bf16)x;
  return __builtin_bit_cast(unsigned short, b);
}

// ---------------- weight conversion f32 -> bf16 ----------------------------
__global__ void convert_weights_kernel(const float* __restrict__ Wih,
                                       const float* __restrict__ Whh,
                                       bf16* __restrict__ wih_bf,
                                       bf16* __restrict__ whh_bf) {
  size_t i = (size_t)blockIdx.x * blockDim.x + threadIdx.x;
  const size_t nwih = (size_t)NLAYERS * 2 * G4H * EMB;
  const size_t nwhh = (size_t)NLAYERS * 2 * G4H * HID;
  if (i < nwih) wih_bf[i] = (bf16)Wih[i];
  if (i < nwhh) whh_bf[i] = (bf16)Whh[i];
}

// ---------------- embedding gather -> bf16 ---------------------------------
__global__ void embed_kernel(const int* __restrict__ words,
                             const float* __restrict__ emb,
                             bf16* __restrict__ x) {
  int token = blockIdx.x;                 // t*BATCH + b
  int w = words[token];
  for (int e = threadIdx.x; e < EMB; e += blockDim.x)
    x[(size_t)token * EMB + e] = (bf16)emb[(size_t)w * EMB + e];
}

// ---------------- input-projection GEMM: C[M,1024] = X[M,512] @ W[1024,512]^T
// BLOCK 64x128, 8 waves; LDS-staged panels (async loads when available).
__global__ __launch_bounds__(256) void gemm_xw_kernel(const bf16* __restrict__ X,
                                                      const bf16* __restrict__ W,
                                                      float* __restrict__ C) {
  __shared__ __attribute__((aligned(16))) unsigned short As[64 * 32];
  __shared__ __attribute__((aligned(16))) unsigned short Bs[128 * 32];
  const int K = EMB, N = G4H;
  const int m0 = blockIdx.x * 64;
  const int n0 = blockIdx.y * 128;
  const int tid = threadIdx.x;
  const int wave = tid >> 5;
  const int mi = wave & 3;   // M sub-tile
  const int nj = wave >> 2;  // N half (64 cols -> 4 wmma tiles)
  const int arow = tid >> 2, ach = tid & 3;   // A: 64 rows x 4x16B
  const int brow = tid >> 1, bch = tid & 1;   // B: 128 rows x 2x32B
  const unsigned short* Xu = (const unsigned short*)X;
  const unsigned short* Wu = (const unsigned short*)W;

  v8f zero = {};
  v8f acc[4];
  for (int i = 0; i < 4; ++i) acc[i] = zero;

  for (int k0 = 0; k0 < K; k0 += 32) {
    copy16_g2l(Xu + (size_t)(m0 + arow) * K + k0 + ach * 8,       As + arow * 32 + ach * 8);
    copy16_g2l(Wu + (size_t)(n0 + brow) * K + k0 + bch * 16,      Bs + brow * 32 + bch * 16);
    copy16_g2l(Wu + (size_t)(n0 + brow) * K + k0 + bch * 16 + 8,  Bs + brow * 32 + bch * 16 + 8);
    async_fence_wait();
    __syncthreads();
    v16bf a = frag_a(As, mi * 16, 0, 32);
#pragma unroll
    for (int i = 0; i < 4; ++i) {
      v16bf b = frag_b(Bs, nj * 64 + i * 16, 0, 32);
      acc[i] = __builtin_amdgcn_wmma_f32_16x16x32_bf16(false, a, false, b,
                                                       (short)0, acc[i], false, false);
    }
    __syncthreads();
  }
  const int lane = tid & 31;
  const int col  = lane & 15;
  const int moff = (lane & 16) ? 8 : 0;
#pragma unroll
  for (int i = 0; i < 4; ++i) {
    int n = n0 + nj * 64 + i * 16 + col;
#pragma unroll
    for (int r = 0; r < 8; ++r)
      C[(size_t)(m0 + mi * 16 + moff + r) * N + n] = acc[i][r];
  }
}

// ---------------- persistent bidirectional LSTM recurrence -----------------
// grid = 2 blocks (fwd/bwd), 1024 threads = 32 waves.
// Per step: gates[64,1024] = h_bf16[64,256] @ Whh^T  (WMMA), + gih[t] + bias.
// Wave w: batch tile (w&3), hidden tiles {2*(w>>2), 2*(w>>2)+1} across all 4
// gates -> i/f/g/o for one (batch,hidden) share lane/VGPR slots; c in regs.
__global__ __launch_bounds__(1024) void lstm_layer_kernel(
    const float* __restrict__ gih_f, const float* __restrict__ gih_b,
    const bf16* __restrict__ Whh2,   // [2][1024][256] bf16 for this layer
    const float* __restrict__ bih2,  // [2][1024]
    const float* __restrict__ bhh2,  // [2][1024]
    bf16*  __restrict__ xnext,       // [SEQ*BATCH][512]: next-layer input
    float* __restrict__ hout)        // [SEQ*BATCH][512]: f32 for fc
{
  const int dir = blockIdx.x;
  const float* gih = dir ? gih_b : gih_f;
  const unsigned short* W = (const unsigned short*)(Whh2 + (size_t)dir * G4H * HID);
  const float* bi = bih2 + dir * G4H;
  const float* bh = bhh2 + dir * G4H;

  __shared__ __attribute__((aligned(16))) unsigned short h_lds[BATCH * HID]; // bf16 h_{t-1}
  __shared__ float bias[G4H];

  const int tid = threadIdx.x;
  bias[tid] = bi[tid] + bh[tid];
#pragma unroll
  for (int i = 0; i < 16; ++i) h_lds[tid * 16 + i] = 0;  // h_0 = 0

  const int wave = tid >> 5, lane = tid & 31;
  const int mbase = (wave & 3) * 16;        // batch rows 16
  const int g     = wave >> 2;              // hidden group 0..7
  const int col   = lane & 15;
  const int moff  = (lane & 16) ? 8 : 0;

  float c_reg[2][8];                        // cell state, in registers
#pragma unroll
  for (int q = 0; q < 2; ++q)
#pragma unroll
    for (int r = 0; r < 8; ++r) c_reg[q][r] = 0.0f;

  __syncthreads();

  for (int t = 0; t < SEQ; ++t) {
    const int tt = dir ? (SEQ - 1 - t) : t;

    v8f zero = {};
    v8f acc[2][4];
#pragma unroll
    for (int q = 0; q < 2; ++q)
#pragma unroll
      for (int gt = 0; gt < 4; ++gt) acc[q][gt] = zero;

#pragma unroll
    for (int kb = 0; kb < 8; ++kb) {
      v16bf a = frag_a(h_lds, mbase, kb * 32, HID);
#pragma unroll
      for (int q = 0; q < 2; ++q) {
        const int ht = 2 * g + q;
#pragma unroll
        for (int gt = 0; gt < 4; ++gt) {
          v16bf b = frag_b(W, gt * HID + ht * 16, kb * 32, HID);
          acc[q][gt] = __builtin_amdgcn_wmma_f32_16x16x32_bf16(
              false, a, false, b, (short)0, acc[q][gt], false, false);
        }
      }
    }
    __syncthreads();  // everyone finished reading h_{t-1}

#pragma unroll
    for (int q = 0; q < 2; ++q) {
      const int hcol = (2 * g + q) * 16 + col;   // hidden index 0..255
#pragma unroll
      for (int r = 0; r < 8; ++r) {
        const int m = mbase + moff + r;          // batch index
        const float* grow = gih + ((size_t)(tt * BATCH + m)) * G4H;
        float gi = acc[q][0][r] + grow[0 * HID + hcol] + bias[0 * HID + hcol];
        float gf = acc[q][1][r] + grow[1 * HID + hcol] + bias[1 * HID + hcol];
        float gg = acc[q][2][r] + grow[2 * HID + hcol] + bias[2 * HID + hcol];
        float go = acc[q][3][r] + grow[3 * HID + hcol] + bias[3 * HID + hcol];
        float c2 = sigf(gf) * c_reg[q][r] + sigf(gi) * tanhf(gg);
        float h2 = sigf(go) * tanhf(c2);
        c_reg[q][r] = c2;
        h_lds[m * HID + hcol] = f2bf_bits(h2);
        size_t o = ((size_t)(tt * BATCH + m)) * IN2 + dir * HID + hcol;
        hout[o]  = h2;
        xnext[o] = (bf16)h2;
      }
    }
    __syncthreads();  // h_t visible before next step's reads
  }
}

// ---------------- fc emissions: em = h @ fcW^T + fcb ----------------------
__global__ void fc_kernel(const float* __restrict__ h, const float* __restrict__ Wt,
                          const float* __restrict__ bt, float* __restrict__ em) {
  int idx = blockIdx.x * blockDim.x + threadIdx.x;
  if (idx >= SEQ * BATCH * NTAGS) return;
  int token = idx / NTAGS, tag = idx % NTAGS;
  const float* hr = h + (size_t)token * IN2;
  const float* wr = Wt + (size_t)tag * IN2;
  float acc = bt[tag];
  for (int k = 0; k < IN2; ++k) acc += hr[k] * wr[k];
  em[idx] = acc;
}

// ---------------- CRF forward (logZ) + numerator ---------------------------
__global__ __launch_bounds__(64) void crf_forward_kernel(
    const float* __restrict__ em, const int* __restrict__ words,
    const int* __restrict__ tags, const float* __restrict__ startt,
    const float* __restrict__ endt, const float* __restrict__ trans,
    float* __restrict__ res) {
  const int b = blockIdx.x, j = threadIdx.x;
  __shared__ float alpha[NTAGS], alpha2[NTAGS];
  __shared__ float tr[NTAGS * NTAGS];
  for (int i = j; i < NTAGS * NTAGS; i += 64) tr[i] = trans[i];
  if (j < NTAGS) alpha[j] = startt[j] + em[(size_t)b * NTAGS + j];
  __syncthreads();
  for (int t = 1; t < SEQ; ++t) {
    bool m = words[t * BATCH + b] != 0;
    if (j < NTAGS) {
      float mx = -3.0e38f;
      for (int i = 0; i < NTAGS; ++i) mx = fmaxf(mx, alpha[i] + tr[i * NTAGS + j]);
      float s = 0.f;
      for (int i = 0; i < NTAGS; ++i) s += __expf(alpha[i] + tr[i * NTAGS + j] - mx);
      float nxt = mx + __logf(s) + em[((size_t)t * BATCH + b) * NTAGS + j];
      alpha2[j] = m ? nxt : alpha[j];
    }
    __syncthreads();
    if (j < NTAGS) alpha[j] = alpha2[j];
    __syncthreads();
  }
  if (j == 0) {
    float mx = -3.0e38f;
    for (int i = 0; i < NTAGS; ++i) mx = fmaxf(mx, alpha[i] + endt[i]);
    float s = 0.f;
    for (int i = 0; i < NTAGS; ++i) s += __expf(alpha[i] + endt[i] - mx);
    float logZ = mx + __logf(s);
    int tg = tags[b];
    float num = startt[tg] + em[(size_t)b * NTAGS + tg];
    int prev = tg, cnt = (words[b] != 0) ? 1 : 0;
    for (int t = 1; t < SEQ; ++t) {
      int cur = tags[t * BATCH + b];
      if (words[t * BATCH + b] != 0) {
        num += tr[prev * NTAGS + cur] + em[((size_t)t * BATCH + b) * NTAGS + cur];
        cnt++;
      }
      prev = cur;
    }
    int li = cnt - 1; if (li < 0) li = 0;
    num += endt[tags[li * BATCH + b]];
    res[b] = num - logZ;
  }
}

__global__ __launch_bounds__(64) void loss_kernel(const float* __restrict__ res,
                                                  float* __restrict__ out) {
  __shared__ float s[BATCH];
  s[threadIdx.x] = res[threadIdx.x];
  __syncthreads();
  if (threadIdx.x == 0) {
    float a = 0.f;
    for (int i = 0; i < BATCH; ++i) a += s[i];
    out[0] = -a;
  }
}

// ---------------- Viterbi --------------------------------------------------
__global__ __launch_bounds__(64) void viterbi_kernel(
    const float* __restrict__ em, const int* __restrict__ words,
    const float* __restrict__ startt, const float* __restrict__ endt,
    const float* __restrict__ trans, unsigned char* __restrict__ hist,
    float* __restrict__ seq_out) {
  const int b = blockIdx.x, j = threadIdx.x;
  __shared__ float sc[NTAGS], sc2[NTAGS];
  __shared__ float tr[NTAGS * NTAGS];
  for (int i = j; i < NTAGS * NTAGS; i += 64) tr[i] = trans[i];
  if (j < NTAGS) sc[j] = startt[j] + em[(size_t)b * NTAGS + j];
  __syncthreads();
  for (int t = 1; t < SEQ; ++t) {
    bool m = words[t * BATCH + b] != 0;
    if (j < NTAGS) {
      float best = -3.0e38f; int bi = 0;
      for (int i = 0; i < NTAGS; ++i) {
        float v = sc[i] + tr[i * NTAGS + j];
        if (v > best) { best = v; bi = i; }
      }
      hist[((size_t)(t - 1) * BATCH + b) * NTAGS + j] = (unsigned char)bi;
      sc2[j] = m ? (best + em[((size_t)t * BATCH + b) * NTAGS + j]) : sc[j];
    }
    __syncthreads();
    if (j < NTAGS) sc[j] = sc2[j];
    __syncthreads();
  }
  if (j == 0) {
    float best = -3.0e38f; int cur = 0;
    for (int i = 0; i < NTAGS; ++i) {
      float v = sc[i] + endt[i];
      if (v > best) { best = v; cur = i; }
    }
    int seq[SEQ];
    seq[SEQ - 1] = cur;
    for (int t = SEQ - 2; t >= 0; --t) {
      bool m = words[(t + 1) * BATCH + b] != 0;
      int prev = m ? (int)hist[((size_t)t * BATCH + b) * NTAGS + cur] : cur;
      seq[t] = prev; cur = prev;
    }
    for (int t = 0; t < SEQ; ++t) {
      bool m = words[t * BATCH + b] != 0;
      seq_out[t * BATCH + b] = m ? (float)seq[t] : 0.0f;
    }
  }
}

// ---------------- host launcher --------------------------------------------
extern "C" void kernel_launch(void* const* d_in, const int* in_sizes, int n_in,
                              void* d_out, int out_size, void* d_ws, size_t ws_size,
                              hipStream_t stream) {
  (void)in_sizes; (void)n_in; (void)out_size; (void)ws_size;
  const int*   words = (const int*)d_in[0];
  const int*   tags  = (const int*)d_in[1];
  const float* emb   = (const float*)d_in[2];
  const float* Wih   = (const float*)d_in[3];
  const float* Whh   = (const float*)d_in[4];
  const float* bih   = (const float*)d_in[5];
  const float* bhh   = (const float*)d_in[6];
  const float* fcW   = (const float*)d_in[7];
  const float* fcb   = (const float*)d_in[8];
  const float* stt   = (const float*)d_in[9];
  const float* ent   = (const float*)d_in[10];
  const float* trn   = (const float*)d_in[11];

  char* ws = (char*)d_ws;
  size_t off = 0;
  auto alloc = [&](size_t bytes) -> void* {
    void* p = (void*)(ws + off);
    off += (bytes + 255) & ~(size_t)255;
    return p;
  };
  bf16*  wih_bf = (bf16*)alloc((size_t)NLAYERS * 2 * G4H * EMB * sizeof(bf16)); // 4 MB
  bf16*  whh_bf = (bf16*)alloc((size_t)NLAYERS * 2 * G4H * HID * sizeof(bf16)); // 2 MB
  bf16*  xa     = (bf16*)alloc((size_t)SEQ * BATCH * IN2 * sizeof(bf16));       // 16 MB
  bf16*  xb     = (bf16*)alloc((size_t)SEQ * BATCH * IN2 * sizeof(bf16));       // 16 MB
  float* h32    = (float*)alloc((size_t)SEQ * BATCH * IN2 * sizeof(float));     // 33 MB
  float* gihf   = (float*)alloc((size_t)SEQ * BATCH * G4H * sizeof(float));     // 67 MB
  float* gihb   = (float*)alloc((size_t)SEQ * BATCH * G4H * sizeof(float));     // 67 MB
  float* emis   = (float*)alloc((size_t)SEQ * BATCH * NTAGS * sizeof(float));   // 3.3 MB
  float* res    = (float*)alloc(BATCH * sizeof(float));
  unsigned char* hist = (unsigned char*)alloc((size_t)(SEQ - 1) * BATCH * NTAGS);

  convert_weights_kernel<<<8192, 256, 0, stream>>>(Wih, Whh, wih_bf, whh_bf);
  embed_kernel<<<SEQ * BATCH, 128, 0, stream>>>(words, emb, xa);

  for (int l = 0; l < NLAYERS; ++l) {
    const bf16* xin  = (l == 0) ? xa : xb;
    bf16*       xout = (l == 0) ? xb : xa;
    const bf16* wl = wih_bf + (size_t)l * 2 * G4H * EMB;
    dim3 grid(SEQ * BATCH / 64, G4H / 128);
    gemm_xw_kernel<<<grid, 256, 0, stream>>>(xin, wl, gihf);
    gemm_xw_kernel<<<grid, 256, 0, stream>>>(xin, wl + (size_t)G4H * EMB, gihb);
    lstm_layer_kernel<<<2, 1024, 0, stream>>>(
        gihf, gihb, whh_bf + (size_t)l * 2 * G4H * HID,
        bih + (size_t)l * 2 * G4H, bhh + (size_t)l * 2 * G4H, xout, h32);
  }

  fc_kernel<<<(SEQ * BATCH * NTAGS + 255) / 256, 256, 0, stream>>>(h32, fcW, fcb, emis);
  crf_forward_kernel<<<BATCH, 64, 0, stream>>>(emis, words, tags, stt, ent, trn, res);
  loss_kernel<<<1, 64, 0, stream>>>(res, (float*)d_out);
  viterbi_kernel<<<BATCH, 64, 0, stream>>>(emis, words, stt, ent, trn, hist,
                                           (float*)d_out + 1);
}
